// SACBLoss_65936337928198
// MI455X (gfx1250) — matrix operand order repaired and checked by reading, not compiled, not run on hardware
//
#include <hip/hip_runtime.h>
#include <math.h>

// CDNA5 / gfx1250: wave32, WMMA 16x16x32 f16 -> f32
typedef __attribute__((ext_vector_type(16))) _Float16 v16h;
typedef __attribute__((ext_vector_type(8)))  float    v8f;

static constexpr int kD             = 256;  // feature dim
static constexpr int kK             = 16;   // neighbors per point
static constexpr int kWavesPerBlock = 8;    // 256 threads = 8 wave32
static constexpr int kBlockThreads  = 256;
static constexpr int kIgnore        = -1;

template <int O>
__device__ __forceinline__ void pack8(v16h& v, float4 p, float4 q) {
  v[O + 0] = (_Float16)p.x;  v[O + 1] = (_Float16)p.y;
  v[O + 2] = (_Float16)p.z;  v[O + 3] = (_Float16)p.w;
  v[O + 4] = (_Float16)q.x;  v[O + 5] = (_Float16)q.y;
  v[O + 6] = (_Float16)q.z;  v[O + 7] = (_Float16)q.w;
}

// ---------------------------------------------------------------------------
// Kernel 1: rnorm[row] = 1 / max(||f_row||_2, 1e-12). One wave per row.
// ---------------------------------------------------------------------------
__global__ void sacb_rnorm_kernel(const float* __restrict__ f,
                                  float* __restrict__ rnorm) {
  const int wave = threadIdx.x >> 5;
  const int lane = threadIdx.x & 31;
  const int row  = blockIdx.x * kWavesPerBlock + wave;
  const float4* p = (const float4*)(f + (size_t)row * kD + lane * 8);
  float4 a = p[0], b = p[1];
  float ss = a.x * a.x + a.y * a.y + a.z * a.z + a.w * a.w +
             b.x * b.x + b.y * b.y + b.z * b.z + b.w * b.w;
#pragma unroll
  for (int off = 16; off >= 1; off >>= 1) ss += __shfl_xor(ss, off, 32);
  if (lane == 0) rnorm[row] = 1.0f / fmaxf(sqrtf(ss), 1e-12f);
}

// ---------------------------------------------------------------------------
// Kernel 2: one wave per point. 16 lanes own the 16 neighbor columns (lane L
// handles neighbor j = L%16; lanes 0-15 feed K 0-7,16-23 of each 32-chunk, lanes
// 16-31 feed K 8-15,24-31). The query row is broadcast into all 16 A rows, so after
// 8 accumulating WMMAs every element of acc row 0 holds a raw dot product.
// Scale by rnorm[n]*rnorm[m] afterwards (dot is linear) — no normalized copy
// of the 64MB table is ever materialized.
// ---------------------------------------------------------------------------
__global__ void sacb_edge_kernel(const float* __restrict__ f,
                                 const int* __restrict__ labels,
                                 const int* __restrict__ idx,
                                 const float* __restrict__ rnorm,
                                 float4* __restrict__ partials) {
  __shared__ float4 red[kWavesPerBlock];
  const int  wave  = threadIdx.x >> 5;
  const int  lane  = threadIdx.x & 31;
  const int  point = blockIdx.x * kWavesPerBlock + wave;
  const int  j     = lane & 15;
  const bool hi    = lane >= 16;

  const int  m      = idx[point * kK + j];
  const bool knn_ok = (m != kIgnore);
  const int  mm     = knn_ok ? m : 0;   // branchless safe gather index

  const float* frow_n = f + (size_t)point * kD;
  const float* frow_m = f + (size_t)mm * kD;
  const int base_a = hi ? 8 : 0;    // A 16x32 f16 lane-group K split (0-7|8-15)
  const int base_b = hi ? 8 : 0;    // B 32x16 f16 lane-group K split (0-7|8-15)

  __builtin_prefetch(frow_m + base_b, 0, 0);  // global_prefetch_b8 on gather row

  v8f acc = {};
#pragma unroll
  for (int c = 0; c < kD / 32; ++c) {
    // A chunk: query row f[n][32c .. 32c+31], lane-group-dependent K slices.
    const int ka = c * 32 + base_a;
    float4 a0 = *(const float4*)(frow_n + ka);
    float4 a1 = *(const float4*)(frow_n + ka + 4);
    float4 a2 = *(const float4*)(frow_n + ka + 16);
    float4 a3 = *(const float4*)(frow_n + ka + 20);
    v16h av;
    pack8<0>(av, a0, a1);   // lo: K 0-7   | hi: K 8-15
    pack8<8>(av, a2, a3);   // lo: K 16-23 | hi: K 24-31

    // B chunk: neighbor j's row, same K slices as A.
    const int kb = c * 32 + base_b;
    float4 b0 = *(const float4*)(frow_m + kb);
    float4 b1 = *(const float4*)(frow_m + kb + 4);
    float4 b2 = *(const float4*)(frow_m + kb + 16);
    float4 b3 = *(const float4*)(frow_m + kb + 20);
    v16h bv;
    pack8<0>(bv, b0, b1);
    pack8<8>(bv, b2, b3);

    // (neg_a, A, neg_b, B, c_mod, C, reuse_a, reuse_b)
    acc = __builtin_amdgcn_wmma_f32_16x16x32_f16(false, av, false, bv,
                                                 (short)0, acc, false, false);
  }

  // Every lane L holds cos for neighbor L%16 (all D rows identical by design).
  const float cosv = acc[0] * rnorm[point] * rnorm[mm];

  const int  ln    = labels[point];
  const int  lm    = labels[mm];
  const bool valid = (!hi) && knn_ok && (ln != kIgnore) && (lm != kIgnore);
  const bool pos   = valid && (ln == lm);
  const bool neg   = valid && (ln != lm);

  float psum = pos ? (1.0f - cosv) : 0.0f;
  float pcnt = pos ? 1.0f : 0.0f;
  float nsum = neg ? fmaxf(cosv - 0.5f, 0.0f) : 0.0f;  // relu(cos - margin)
  float ncnt = neg ? 1.0f : 0.0f;
#pragma unroll
  for (int off = 16; off >= 1; off >>= 1) {
    psum += __shfl_xor(psum, off, 32);
    pcnt += __shfl_xor(pcnt, off, 32);
    nsum += __shfl_xor(nsum, off, 32);
    ncnt += __shfl_xor(ncnt, off, 32);
  }
  if (lane == 0) red[wave] = make_float4(psum, pcnt, nsum, ncnt);
  __syncthreads();
  if (threadIdx.x == 0) {  // fixed-order combine: deterministic
    float4 s = red[0];
#pragma unroll
    for (int w = 1; w < kWavesPerBlock; ++w) {
      s.x += red[w].x; s.y += red[w].y; s.z += red[w].z; s.w += red[w].w;
    }
    partials[blockIdx.x] = s;
  }
}

// ---------------------------------------------------------------------------
// Kernel 3: fixed-tree reduction of per-block partials -> scalar loss.
// ---------------------------------------------------------------------------
__global__ void sacb_finalize_kernel(const float4* __restrict__ partials,
                                     int num_partials,
                                     float* __restrict__ out) {
  __shared__ float4 sd[kBlockThreads];
  const int t = threadIdx.x;
  float4 s = make_float4(0.f, 0.f, 0.f, 0.f);
  for (int i = t; i < num_partials; i += kBlockThreads) {
    float4 p = partials[i];
    s.x += p.x; s.y += p.y; s.z += p.z; s.w += p.w;
  }
  sd[t] = s;
  __syncthreads();
  for (int stride = kBlockThreads / 2; stride >= 1; stride >>= 1) {
    if (t < stride) {
      sd[t].x += sd[t + stride].x; sd[t].y += sd[t + stride].y;
      sd[t].z += sd[t + stride].z; sd[t].w += sd[t + stride].w;
    }
    __syncthreads();
  }
  if (t == 0) {
    const float pos_loss = sd[0].x / fmaxf(sd[0].y, 1.0f);
    const float neg_loss =
        (sd[0].w > 0.0f) ? (sd[0].z / fmaxf(sd[0].w, 1.0f)) : 0.0f;
    out[0] = pos_loss + 0.5f * neg_loss;  // LOSS_WEIGHT = 1.0
  }
}

// ---------------------------------------------------------------------------
// inputs: d_in[0]=features f32 [N,256], d_in[1]=labels int [N],
//         d_in[2]=idx int [N,16]; d_out: 1 float.
// ws layout: [0, blocks) float4 partials | then N floats rnorm. (~384 KB)
// ---------------------------------------------------------------------------
extern "C" void kernel_launch(void* const* d_in, const int* in_sizes, int n_in,
                              void* d_out, int out_size, void* d_ws,
                              size_t ws_size, hipStream_t stream) {
  const float* f      = (const float*)d_in[0];
  const int*   labels = (const int*)d_in[1];
  const int*   idx    = (const int*)d_in[2];
  float*       out    = (float*)d_out;

  const int n      = in_sizes[1];            // 65536
  const int blocks = n / kWavesPerBlock;     // 8192

  float4* partials = (float4*)d_ws;
  float*  rnorm    = (float*)d_ws + (size_t)blocks * 4;

  sacb_rnorm_kernel<<<blocks, kBlockThreads, 0, stream>>>(f, rnorm);
  sacb_edge_kernel<<<blocks, kBlockThreads, 0, stream>>>(f, labels, idx, rnorm,
                                                         partials);
  sacb_finalize_kernel<<<1, kBlockThreads, 0, stream>>>(partials, blocks, out);
}
